// SelectiveCell_67843303408278
// MI455X (gfx1250) — matrix operand order repaired and checked by reading, not compiled, not run on hardware
//
#include <hip/hip_runtime.h>

typedef __attribute__((ext_vector_type(16))) __bf16  v16bf;
typedef __attribute__((ext_vector_type(8)))  float   v8f;

namespace {
constexpr int    B_    = 8;
constexpr int    T_    = 2048;
constexpr int    C_    = 64;
constexpr int    H_    = 256;
constexpr int    HP_   = 64;                 // H / POOL
constexpr int    KTOT  = C_ * H_;            // 16384
constexpr int    NT    = 16;                 // N tiles of 16 cols
constexpr int    KG    = 4;                  // K groups
constexpr int    NWG   = NT * KG;            // 64 persistent workgroups
constexpr int    NTHR  = 256;                // 8 waves (wave32)
constexpr int    NWAVE = 8;
constexpr int    CHUNKS = KTOT / KG / (NWAVE * 32); // 16 k32-chunks per wave
// workspace layout (bytes)
constexpr size_t WS_WB = 0;                                     // bf16 swizzled W_wt: 8 MiB
constexpr size_t SZ_WB = (size_t)KTOT * H_ * 2;
constexpr size_t WS_FP = SZ_WB;                                 // f_parts[2][KG*NT*256] f32
constexpr size_t SZ_FP = (size_t)2 * KG * NT * 256 * 4;
constexpr size_t WS_BG = WS_FP + SZ_FP;                         // bf_g[2][B][H] f32
constexpr size_t SZ_BG = (size_t)2 * B_ * H_ * 4;
constexpr size_t WS_FL = WS_BG + SZ_BG;                         // flags[NWG] u32
}

// ---------------------------------------------------------------------------
// Prep: convert W_wt (fp32 [C][H][H]) into the per-lane WMMA-B bf16 layout:
//   Wb[((kc*16 + nt)*32 + lane)*16 + v]  with  n = nt*16 + (lane&15),
//   k = kc*32 + (lane<16 ? 0 : 16) + v,  element = W_wt[k*H + n].
// Also zeroes the distributed barrier flags (deterministic across replays).
// ---------------------------------------------------------------------------
__global__ __launch_bounds__(256) void selcell_prep(const float* __restrict__ W_wt,
                                                    __bf16* __restrict__ Wb,
                                                    unsigned* __restrict__ flags) {
  size_t i = (size_t)blockIdx.x * 256 + threadIdx.x;
  if (i < NWG) flags[i] = 0u;
  if (i >= (size_t)KTOT * H_) return;
  int v    = (int)(i & 15);
  int lane = (int)((i >> 4) & 31);
  int nt   = (int)((i >> 9) & 15);
  int kc   = (int)(i >> 13);
  int n    = nt * 16 + (lane & 15);
  int k    = kc * 32 + ((lane >= 16) ? 16 : 0) + v;
  Wb[i] = (__bf16)W_wt[(size_t)k * H_ + n];
}

// Distributed-flag grid barrier: parallel release stores, one wave polls.
__device__ __forceinline__ void grid_barrier(unsigned* flags, unsigned epoch,
                                             int wg, int tid) {
  __threadfence();                       // make this thread's stores visible
  __syncthreads();
  if (tid == 0)
    __hip_atomic_store(&flags[wg], epoch, __ATOMIC_RELEASE, __HIP_MEMORY_SCOPE_AGENT);
  if (tid < 32) {
    while (__hip_atomic_load(&flags[tid], __ATOMIC_RELAXED,
                             __HIP_MEMORY_SCOPE_AGENT) < epoch ||
           __hip_atomic_load(&flags[tid + 32], __ATOMIC_RELAXED,
                             __HIP_MEMORY_SCOPE_AGENT) < epoch)
      __builtin_amdgcn_s_sleep(1);
  }
  __threadfence();                       // acquire side
  __syncthreads();
}

// ---------------------------------------------------------------------------
// Persistent recurrent cell. Grid = 64 WGs (16 n-tiles x 4 k-groups), 256 thr.
// ONE grid barrier per time step.
// ---------------------------------------------------------------------------
__global__ __launch_bounds__(NTHR) void selcell_main(
    const float* __restrict__ x,    const float* __restrict__ W_h,
    const float* __restrict__ b_h,  const float* __restrict__ b_wt,
    const float* __restrict__ W_bt, const float* __restrict__ b_bt,
    const float* __restrict__ W_e,  const float* __restrict__ b_e,
    const __bf16* __restrict__ Wb,  float* __restrict__ f_parts,
    float* __restrict__ bf_g,       unsigned* __restrict__ flags,
    float* __restrict__ out) {
  __shared__ float W_h_lds[C_][H_];      // 64 KB, staged once (CDNA5: 320KB LDS)
  __shared__ float We_t[HP_][H_];        // 64 KB, W_e transposed, staged once
  __shared__ float x_s[B_][C_ + 4];
  __shared__ float pf_s[B_][HP_];
  __shared__ float h_s[B_][H_];          // also reused as f-store in finalize
  __shared__ float ph_s[B_][HP_];
  __shared__ float he_s[B_][H_ + 4];     // +4 pad for A-frag bank spread
  __shared__ float red[NWAVE * 256];

  const int tid  = threadIdx.x;
  const int wg   = blockIdx.x;
  const int nt   = wg & (NT - 1);
  const int kg   = wg >> 4;
  const int lane = tid & 31;
  const int wave = tid >> 5;
  const int jcol = tid;                  // hidden column owned by this thread

  // ---- one-time staging of step-invariant weights into LDS ----
  for (int idx = tid; idx < C_ * H_; idx += NTHR)
    W_h_lds[idx >> 8][idx & 255] = W_h[idx];
  for (int idx = tid; idx < HP_ * H_; idx += NTHR) {
    int p = idx >> 8, j = idx & 255;
    We_t[p][j] = W_e[j * HP_ + p];       // transpose: lane-stride-1 reads later
  }

  for (int t = 0; t < T_; ++t) {
    const int pcur = t & 1, pprev = (t ^ 1) & 1;

    // ---------- finalize step t-1 (redundant in every WG): f -> pf ----------
    if (t > 0) {
      const float* fp = f_parts + (size_t)pprev * (KG * NT * 256);
      const float* bg = bf_g + (size_t)pprev * (B_ * H_);
      const int ntJ = jcol >> 4, cJ = jcol & 15;
#pragma unroll
      for (int b = 0; b < B_; ++b) {
        float s = bg[b * H_ + jcol];
#pragma unroll
        for (int g = 0; g < KG; ++g)
          s += fp[(g * NT + ntJ) * 256 + b * 16 + cJ];
        h_s[b][jcol] = s > 0.f ? s : 0.f;   // f, staged in h_s temporarily
      }
      __syncthreads();
      for (int idx = tid; idx < B_ * HP_; idx += NTHR) {
        int b = idx >> 6, vq = idx & 63;
        const float* fr = &h_s[b][vq * 4];
        pf_s[b][vq] = fmaxf(fmaxf(fr[0], fr[1]), fmaxf(fr[2], fr[3]));
      }
    }
    // stage x_t
    for (int idx = tid; idx < B_ * C_; idx += NTHR) {
      int b = idx >> 6, c = idx & 63;
      x_s[b][c] = x[((size_t)b * T_ + t) * C_ + c];
    }
    __syncthreads();

    // ---------- h = relu(x@W_h + b_h + pf@W_e^T + b_e), 8 batch accs --------
    {
      float acc[B_];
      const float bh = b_h[jcol];
#pragma unroll
      for (int b = 0; b < B_; ++b) acc[b] = bh;
      if (t > 0) {
        const float be = b_e[jcol];
#pragma unroll
        for (int b = 0; b < B_; ++b) acc[b] += be;
        for (int p = 0; p < HP_; ++p) {
          float wv = We_t[p][jcol];
#pragma unroll
          for (int b = 0; b < B_; ++b) acc[b] += pf_s[b][p] * wv;
        }
      }
      for (int c = 0; c < C_; ++c) {
        float wv = W_h_lds[c][jcol];
#pragma unroll
        for (int b = 0; b < B_; ++b) acc[b] += x_s[b][c] * wv;
      }
#pragma unroll
      for (int b = 0; b < B_; ++b) h_s[b][jcol] = acc[b] > 0.f ? acc[b] : 0.f;
    }
    __syncthreads();

    for (int idx = tid; idx < B_ * HP_; idx += NTHR) {
      int b = idx >> 6, vq = idx & 63;
      const float* hr = &h_s[b][vq * 4];
      ph_s[b][vq] = fmaxf(fmaxf(hr[0], hr[1]), fmaxf(hr[2], hr[3]));
    }
    __syncthreads();

    // ---------- he = pool_expand(h) = ph@W_e^T + b_e (the step output) ------
    {
      float acc[B_];
      const float be = b_e[jcol];
#pragma unroll
      for (int b = 0; b < B_; ++b) acc[b] = be;
      for (int p = 0; p < HP_; ++p) {
        float wv = We_t[p][jcol];
#pragma unroll
        for (int b = 0; b < B_; ++b) acc[b] += ph_s[b][p] * wv;
      }
#pragma unroll
      for (int b = 0; b < B_; ++b) he_s[b][jcol] = acc[b];
    }
    __syncthreads();

    if (kg == 1) {                       // write the output slice for this nt
      for (int idx = tid; idx < B_ * 16; idx += NTHR) {
        int b = idx >> 4, jj = idx & 15;
        out[((size_t)b * T_ + t) * H_ + nt * 16 + jj] = he_s[b][nt * 16 + jj];
      }
    } else if (kg == 0 && tid < B_ * 16) {  // bias-ish terms -> global (dbl-buf)
      int b = tid >> 4, jj = tid & 15, col = nt * 16 + jj;
      float acc = b_bt[col];
      for (int c = 0; c < C_; ++c)   acc += x_s[b][c] * W_bt[c * H_ + col];
      for (int hh = 0; hh < H_; ++hh) acc += he_s[b][hh] * b_wt[hh * H_ + col];
      bf_g[(size_t)pcur * (B_ * H_) + b * H_ + col] = acc;
    }

    // ---------- big GEMM: f_pre = (x (x) he) @ W_wt, bf16 WMMA --------------
    // kc0 is a multiple of 16 -> the 16 chunks span exactly two x columns and
    // hhb = (i&7)*32 is a compile-time constant per unrolled iteration.
    {
      const int m    = lane & 15;        // A row (batch), rows 8..15 pad
      const int m2   = (m < B_) ? m : 0;
      const int kc0  = kg * (KTOT / KG / 32) + wave * CHUNKS;
      const int c0   = kc0 >> 3;
      const float xv0 = (m < B_) ? x_s[m2][c0]     : 0.f;  // hoisted: 2 LDS loads
      const float xv1 = (m < B_) ? x_s[m2][c0 + 1] : 0.f;  // instead of 16
      const int kbase = (lane < 16) ? 0 : 16;
      const float* herow = &he_s[m2][kbase];
      const __bf16* bbase = Wb + (((size_t)kc0 * NT + nt) * 32 + lane) * 16;
      v8f cacc0 = {}, cacc1 = {};        // two chains: independent WMMAs overlap
#pragma unroll
      for (int i = 0; i < CHUNKS; ++i) {
        const float xv  = (i < 8) ? xv0 : xv1;
        const int   hhb = (i & 7) * 32;  // constant offset after unroll
        v16bf bfrag = *(const v16bf*)(bbase + (size_t)i * (NT * 32 * 16));
        v16bf afrag;
#pragma unroll
        for (int v = 0; v < 16; ++v) {
          int koff = hhb + v;   // contiguous K order (matches Wb)
          afrag[v] = (__bf16)(xv * herow[koff]);
        }
        if (i & 1)
          cacc1 = __builtin_amdgcn_wmma_f32_16x16x32_bf16(
              false, afrag, false, bfrag, (short)0, cacc1, false, false);
        else
          cacc0 = __builtin_amdgcn_wmma_f32_16x16x32_bf16(
              false, afrag, false, bfrag, (short)0, cacc0, false, false);
      }
      const int row0 = (lane < 16) ? 0 : 8;     // C layout
      const int colc = lane & 15;
#pragma unroll
      for (int r = 0; r < 8; ++r)
        red[wave * 256 + (row0 + r) * 16 + colc] = cacc0[r] + cacc1[r];
    }
    __syncthreads();
    {
      float s = 0.f;
#pragma unroll
      for (int w = 0; w < NWAVE; ++w) s += red[w * 256 + tid];
      f_parts[(size_t)pcur * (KG * NT * 256) + ((size_t)kg * NT + nt) * 256 + tid] = s;
    }

    grid_barrier(flags, (unsigned)(t + 1), wg, tid);  // ONE barrier per step
  }
}

extern "C" void kernel_launch(void* const* d_in, const int* in_sizes, int n_in,
                              void* d_out, int out_size, void* d_ws, size_t ws_size,
                              hipStream_t stream) {
  const float* xin  = (const float*)d_in[0];
  const float* W_h  = (const float*)d_in[1];
  const float* b_h  = (const float*)d_in[2];
  const float* W_wt = (const float*)d_in[3];
  const float* b_wt = (const float*)d_in[4];
  const float* W_bt = (const float*)d_in[5];
  const float* b_bt = (const float*)d_in[6];
  const float* W_e  = (const float*)d_in[7];
  const float* b_e  = (const float*)d_in[8];
  float* out = (float*)d_out;

  char* ws = (char*)d_ws;
  __bf16*   Wb      = (__bf16*)(ws + WS_WB);
  float*    f_parts = (float*)(ws + WS_FP);
  float*    bf_g    = (float*)(ws + WS_BG);
  unsigned* flags   = (unsigned*)(ws + WS_FL);

  const int prep_blocks = (KTOT * H_ + 255) / 256;   // 16384
  selcell_prep<<<prep_blocks, 256, 0, stream>>>(W_wt, Wb, flags);
  selcell_main<<<NWG, NTHR, 0, stream>>>(xin, W_h, b_h, b_wt, W_bt, b_bt,
                                         W_e, b_e, Wb, f_parts, bf_g, flags, out);
}
